// CausalSelectiveSelfAttentionWithMemoryPenalty_81355270521195
// MI455X (gfx1250) — compile-verified, matrix operand order checked by reading, not run on hardware
//
#include <hip/hip_runtime.h>
#include <hip/hip_bf16.h>

// ---------------- problem constants ----------------
#define B_   2
#define T_   2048
#define C_   768
#define H_   12
#define D_   64
#define HD_  768      // H_*D_
#define N3_  2304     // 3*HD_
#define SCALE_ 0.125f // 1/sqrt(64)
#define CH_  16       // scan chunks
#define CR_  (T_/CH_) // 128 rows per chunk

typedef __attribute__((ext_vector_type(16))) _Float16 v16h;
typedef __attribute__((ext_vector_type(8)))  _Float16 v8h;
typedef __attribute__((ext_vector_type(8)))  float    v8f;

__device__ __forceinline__ v8f zero8() {
  v8f z = {0.f,0.f,0.f,0.f,0.f,0.f,0.f,0.f};
  return z;
}

// 16-half fragment: halves 0..7 from p[0..7], halves 8..15 from p[16..23]
// (per-lane kbase = 8*(lane>=16) handled by caller) -> two b128 loads.
__device__ __forceinline__ v16h load_frag(const _Float16* p) {
  v8h lo = *(const v8h*)(p);
  v8h hi = *(const v8h*)(p + 16);
  return __builtin_shufflevector(lo, hi, 0,1,2,3,4,5,6,7,8,9,10,11,12,13,14,15);
}

__device__ __forceinline__ v8f wmma_f16(v16h a, v16h b, v8f c) {
  return __builtin_amdgcn_wmma_f32_16x16x32_f16(false, a, false, b, (short)0, c, false, false);
}

// 8 WMMAs for a 32x64 tile (2 row-tiles x 4 col-tiles)
#define GEMM8(A0, A1, Bq0, Bq1, Bq2, Bq3)                                   \
  do {                                                                      \
    acc[0] = wmma_f16(A0, Bq0, acc[0]); acc[4] = wmma_f16(A1, Bq0, acc[4]); \
    acc[1] = wmma_f16(A0, Bq1, acc[1]); acc[5] = wmma_f16(A1, Bq1, acc[5]); \
    acc[2] = wmma_f16(A0, Bq2, acc[2]); acc[6] = wmma_f16(A1, Bq2, acc[6]); \
    acc[3] = wmma_f16(A0, Bq3, acc[3]); acc[7] = wmma_f16(A1, Bq3, acc[7]); \
  } while (0)

// ---------------- prep: f32 -> f16 conversions / transposes ----------------
__global__ __launch_bounds__(256) void cvt_f16_kernel(const float* __restrict__ src,
                                                      _Float16* __restrict__ dst, int n) {
  int i = blockIdx.x * 256 + threadIdx.x;
  if (i < n) dst[i] = (_Float16)src[i];
}

// dst[n*rows + r] = src[r*cols + n]   (transpose + convert)
__global__ __launch_bounds__(256) void cvt_t_kernel(const float* __restrict__ src,
                                                    _Float16* __restrict__ dst,
                                                    int rows, int cols) {
  int i = blockIdx.x * 256 + threadIdx.x;
  int n = rows * cols;
  if (i >= n) return;
  int nn = i / rows;
  int rr = i - nn * rows;
  dst[(size_t)nn * rows + rr] = (_Float16)src[(size_t)rr * cols + nn];
}

// ---------------- qkv GEMM: X[4096,768] @ WattnT + b -> q,k,vT (f16); 32x64 tile, ping-pong ----------------
__global__ __launch_bounds__(256) void qkv_gemm_kernel(const _Float16* __restrict__ X,
                                                       const _Float16* __restrict__ WT,
                                                       const float* __restrict__ bias,
                                                       _Float16* __restrict__ Q,
                                                       _Float16* __restrict__ K,
                                                       _Float16* __restrict__ VT) {
  const int NT = N3_ / 64;                       // 36 col-tiles of 64
  int gwave = (blockIdx.x * 256 + threadIdx.x) >> 5;
  int lane  = threadIdx.x & 31;
  int mt = gwave / NT, nt = gwave % NT;
  if (mt >= (B_ * T_) / 32) return;
  int m0 = mt * 32, n0 = nt * 64;
  int laneN = lane & 15;
  int kb    = (lane >> 4) << 3;                  // 0 or 8
  const _Float16* pa0 = X + (size_t)(m0 + laneN) * C_ + kb;
  const _Float16* pa1 = X + (size_t)(m0 + 16 + laneN) * C_ + kb;
  const _Float16* pb0 = WT + (size_t)(n0 + 0  + laneN) * C_ + kb;
  const _Float16* pb1 = WT + (size_t)(n0 + 16 + laneN) * C_ + kb;
  const _Float16* pb2 = WT + (size_t)(n0 + 32 + laneN) * C_ + kb;
  const _Float16* pb3 = WT + (size_t)(n0 + 48 + laneN) * C_ + kb;
  v8f acc[8];
  #pragma unroll
  for (int i = 0; i < 8; ++i) acc[i] = zero8();
  // ping-pong fragment sets: X-set preloaded at k=0
  v16h xa0 = load_frag(pa0), xa1 = load_frag(pa1);
  v16h xb0 = load_frag(pb0), xb1 = load_frag(pb1);
  v16h xb2 = load_frag(pb2), xb3 = load_frag(pb3);
  #pragma unroll 1
  for (int k = 0; k < C_; k += 64) {             // 12 trips, no fragment copies
    int k1 = k + 32;
    // speculative prefetch: overshoot past row end is silently dropped (ISA 10.5)
    __builtin_prefetch(pa0 + k + 128, 0, 3);
    __builtin_prefetch(pa1 + k + 128, 0, 3);
    __builtin_prefetch(pb0 + k + 128, 0, 3);
    __builtin_prefetch(pb1 + k + 128, 0, 3);
    __builtin_prefetch(pb2 + k + 128, 0, 3);
    __builtin_prefetch(pb3 + k + 128, 0, 3);
    v16h ya0 = load_frag(pa0 + k1), ya1 = load_frag(pa1 + k1);
    v16h yb0 = load_frag(pb0 + k1), yb1 = load_frag(pb1 + k1);
    v16h yb2 = load_frag(pb2 + k1), yb3 = load_frag(pb3 + k1);
    GEMM8(xa0, xa1, xb0, xb1, xb2, xb3);
    if (k + 64 < C_) {
      int k2 = k + 64;
      xa0 = load_frag(pa0 + k2); xa1 = load_frag(pa1 + k2);
      xb0 = load_frag(pb0 + k2); xb1 = load_frag(pb1 + k2);
      xb2 = load_frag(pb2 + k2); xb3 = load_frag(pb3 + k2);
    }
    GEMM8(ya0, ya1, yb0, yb1, yb2, yb3);
  }
  int rbase = kb;                                // row = r + 8*(lane>=16)
  #pragma unroll
  for (int half = 0; half < 2; ++half) {
    #pragma unroll
    for (int j = 0; j < 4; ++j) {
      int n = n0 + j * 16 + laneN;
      float bn = bias[n];
      int which = n / HD_;
      int hh = (n % HD_) / D_;
      int d  = n % D_;
      #pragma unroll
      for (int r = 0; r < 8; ++r) {
        int m = m0 + half * 16 + r + rbase;
        int b = m / T_, t = m - b * T_;
        _Float16 hv = (_Float16)(acc[half * 4 + j][r] + bn);
        if (which == 0)      Q[(((size_t)b * H_ + hh) * T_ + t) * D_ + d] = hv;
        else if (which == 1) K[(((size_t)b * H_ + hh) * T_ + t) * D_ + d] = hv;
        else                 VT[(((size_t)b * H_ + hh) * D_ + d) * T_ + t] = hv;
      }
    }
  }
}

// ---------------- head-0 masked scores S[B,T,T]; 16x64 wave tile ----------------
__global__ __launch_bounds__(256) void s0_kernel(const _Float16* __restrict__ Q,
                                                 const _Float16* __restrict__ K,
                                                 float* __restrict__ S) {
  const int TT = T_ / 16;                        // 128
  const int GT = TT / 4;                         // 32 groups of 64 cols
  int gwave = (blockIdx.x * 256 + threadIdx.x) >> 5;
  int lane  = threadIdx.x & 31;
  if (gwave >= B_ * TT * GT) return;
  int b  = gwave / (TT * GT);
  int r2 = gwave - b * (TT * GT);
  int tt = r2 / GT, gg = r2 - tt * GT;
  int t0 = tt * 16, sbase = gg * 64;
  int laneN = lane & 15;
  int kb    = (lane >> 4) << 3;
  int rbase = kb;
  float* srow = S + (size_t)b * T_ * T_;
  if (sbase > t0) {                              // whole 16x64 group above diagonal
    #pragma unroll
    for (int st = 0; st < 4; ++st)
      #pragma unroll
      for (int r = 0; r < 8; ++r)
        srow[(size_t)(t0 + r + rbase) * T_ + sbase + st * 16 + laneN] = 0.f;
    return;
  }
  const _Float16* qb  = Q + ((size_t)b * H_) * T_ * D_;   // head 0
  const _Float16* kbp = K + ((size_t)b * H_) * T_ * D_;
  v16h qa0 = load_frag(qb + (size_t)(t0 + laneN) * D_ + kb);
  v16h qa1 = load_frag(qb + (size_t)(t0 + laneN) * D_ + 32 + kb);
  #pragma unroll
  for (int st = 0; st < 4; ++st) {
    int s0 = sbase + st * 16;
    if (s0 > t0) {
      #pragma unroll
      for (int r = 0; r < 8; ++r)
        srow[(size_t)(t0 + r + rbase) * T_ + s0 + laneN] = 0.f;
      continue;
    }
    v16h kb0 = load_frag(kbp + (size_t)(s0 + laneN) * D_ + kb);
    v16h kb1 = load_frag(kbp + (size_t)(s0 + laneN) * D_ + 32 + kb);
    v8f acc = zero8();
    acc = wmma_f16(qa0, kb0, acc);
    acc = wmma_f16(qa1, kb1, acc);
    #pragma unroll
    for (int r = 0; r < 8; ++r) {
      int t = t0 + r + rbase;
      int s = s0 + laneN;
      float v = acc[r] * SCALE_;
      v = (s > 0 && s < t) ? fmaxf(v, 0.f) : 0.f;
      srow[(size_t)t * T_ + s] = v;
    }
  }
}

// ---------------- 3-phase column scan: FF[b,t,s] = sum_{t'<t} S[b,t',s] (in place) ----------------
__global__ __launch_bounds__(256) void scan_phase1_kernel(const float* __restrict__ S,
                                                          float* __restrict__ part) {
  int gid = blockIdx.x * 256 + threadIdx.x;      // B_*CH_*T_ threads
  if (gid >= B_ * CH_ * T_) return;
  int s = gid % T_;
  int rem = gid / T_;
  int c = rem % CH_;
  int b = rem / CH_;
  size_t base = (size_t)b * T_ * T_ + (size_t)(c * CR_) * T_ + s;
  float acc = 0.f;
  for (int t = 0; t < CR_; ++t) acc += S[base + (size_t)t * T_];
  part[gid] = acc;
}

__global__ __launch_bounds__(256) void scan_phase2_kernel(float* __restrict__ part) {
  int gid = blockIdx.x * 256 + threadIdx.x;      // B_*T_ threads
  if (gid >= B_ * T_) return;
  int s = gid % T_;
  int b = gid / T_;
  float run = 0.f;
  #pragma unroll
  for (int c = 0; c < CH_; ++c) {
    size_t idx = ((size_t)b * CH_ + c) * T_ + s;
    float v = part[idx];
    part[idx] = run;
    run += v;
  }
}

__global__ __launch_bounds__(256) void scan_phase3_kernel(float* __restrict__ S,
                                                          const float* __restrict__ part) {
  int gid = blockIdx.x * 256 + threadIdx.x;      // B_*CH_*T_ threads
  if (gid >= B_ * CH_ * T_) return;
  int s = gid % T_;
  int rem = gid / T_;
  int c = rem % CH_;
  int b = rem / CH_;
  size_t base = (size_t)b * T_ * T_ + (size_t)(c * CR_) * T_ + s;
  float run = part[gid];
  for (int t = 0; t < CR_; ++t) {
    size_t idx = base + (size_t)t * T_;
    float v = S[idx];
    S[idx] = run;
    run += v;
  }
}

// ---------------- FF_sum[b,r] = sum_s min(FF[b,r,s], 1.0) ----------------
__global__ __launch_bounds__(256) void ffsum_kernel(const float* __restrict__ FF,
                                                    float* __restrict__ FSUM) {
  int rowid = blockIdx.x;                        // B_*T_ rows
  int b = rowid / T_, r = rowid - b * T_;
  const float* p = FF + ((size_t)b * T_ + r) * T_;
  float acc = 0.f;
  for (int s = threadIdx.x; s < T_; s += 256) acc += fminf(p[s], 1.0f);
  #pragma unroll
  for (int off = 16; off >= 1; off >>= 1) acc += __shfl_xor(acc, off, 32);
  __shared__ float red[8];
  if ((threadIdx.x & 31) == 0) red[threadIdx.x >> 5] = acc;
  __syncthreads();
  if (threadIdx.x == 0) {
    float s = 0.f;
    #pragma unroll
    for (int i = 0; i < 8; ++i) s += red[i];
    FSUM[rowid] = s;
  }
}

// ---------------- M[0,b,t,s] = t - FF_sum[b,s] ----------------
__global__ __launch_bounds__(256) void m_kernel(const float* __restrict__ FSUM,
                                                float* __restrict__ M) {
  size_t gid = (size_t)blockIdx.x * 256 + threadIdx.x;
  if (gid >= (size_t)B_ * T_ * T_) return;
  int s = (int)(gid % T_);
  size_t rem = gid / T_;
  int t = (int)(rem % T_);
  int b = (int)(rem / T_);
  M[gid] = (float)t - FSUM[b * T_ + s];
}

// ---------------- flash attention: softmax(q k^T * scale - FF) @ v -> Yh f16 ----------------
__global__ __launch_bounds__(256) void attn_kernel(const _Float16* __restrict__ Q,
                                                   const _Float16* __restrict__ Kv,
                                                   const _Float16* __restrict__ VT,
                                                   const float* __restrict__ FF,
                                                   _Float16* __restrict__ Yh) {
  __shared__ _Float16 lds_p[8][16 * 32];         // per-wave 16x32 P tile (8 KB)
  int wib  = threadIdx.x >> 5;
  int lane = threadIdx.x & 31;
  int gwave = blockIdx.x * 8 + wib;
  const int TT = T_ / 16;
  if (gwave >= B_ * H_ * TT) return;
  int tt = gwave % TT;
  int bh = gwave / TT;
  int h = bh % H_, b = bh / H_;
  int t0 = tt * 16;
  int laneN = lane & 15;
  int kb    = (lane >> 4) << 3;
  int rbase = kb;
  const _Float16* qb  = Q  + (((size_t)b * H_ + h) * T_) * D_;
  const _Float16* kbp = Kv + (((size_t)b * H_ + h) * T_) * D_;
  const _Float16* vb  = VT + (((size_t)b * H_ + h) * D_) * T_;
  const float* ffb = FF + (size_t)b * T_ * T_;
  v16h qa0 = load_frag(qb + (size_t)(t0 + laneN) * D_ + kb);
  v16h qa1 = load_frag(qb + (size_t)(t0 + laneN) * D_ + 32 + kb);
  v8f o0 = zero8(), o1 = zero8(), o2 = zero8(), o3 = zero8();
  float m_r[8], l_r[8];
  #pragma unroll
  for (int r = 0; r < 8; ++r) { m_r[r] = -1e30f; l_r[r] = 0.f; }
  _Float16* pbuf = lds_p[wib];
  int s_end = t0 + 15;
  for (int sc = 0; sc <= s_end; sc += 32) {
    // speculative prefetch of next chunk's K rows (overshoot dropped by HW)
    __builtin_prefetch(kbp + (size_t)(sc + 32 + laneN) * D_ + kb, 0, 3);
    __builtin_prefetch(kbp + (size_t)(sc + 48 + laneN) * D_ + kb, 0, 3);
    // ---- scores for two 16-col tiles ----
    v8f sa0 = zero8(), sa1 = zero8();
    {
      int srow = sc + laneN;
      v16h k0 = load_frag(kbp + (size_t)srow * D_ + kb);
      v16h k1 = load_frag(kbp + (size_t)srow * D_ + 32 + kb);
      sa0 = wmma_f16(qa0, k0, sa0);
      sa0 = wmma_f16(qa1, k1, sa0);
    }
    if (sc + 16 <= s_end) {
      int srow = sc + 16 + laneN;
      v16h k0 = load_frag(kbp + (size_t)srow * D_ + kb);
      v16h k1 = load_frag(kbp + (size_t)srow * D_ + 32 + kb);
      sa1 = wmma_f16(qa0, k0, sa1);
      sa1 = wmma_f16(qa1, k1, sa1);
    }
    // ---- V fragments: independent of softmax, issue early to overlap VALU ----
    v16h vb0 = load_frag(vb + (size_t)(0  + laneN) * T_ + sc + kb);
    v16h vb1 = load_frag(vb + (size_t)(16 + laneN) * T_ + sc + kb);
    v16h vb2 = load_frag(vb + (size_t)(32 + laneN) * T_ + sc + kb);
    v16h vb3 = load_frag(vb + (size_t)(48 + laneN) * T_ + sc + kb);
    // ---- mask, subtract FF, online softmax ----
    float p0[8], p1[8];
    #pragma unroll
    for (int r = 0; r < 8; ++r) {
      int t = t0 + r + rbase;
      int s0i = sc + laneN;
      int s1i = sc + 16 + laneN;
      p0[r] = (s0i <= t) ? (sa0[r] * SCALE_ - ffb[(size_t)t * T_ + s0i]) : -1e30f;
      p1[r] = (s1i <= t) ? (sa1[r] * SCALE_ - ffb[(size_t)t * T_ + s1i]) : -1e30f;
    }
    #pragma unroll
    for (int r = 0; r < 8; ++r) {
      float mx = fmaxf(p0[r], p1[r]);
      #pragma unroll
      for (int off = 8; off >= 1; off >>= 1) mx = fmaxf(mx, __shfl_xor(mx, off, 32));
      float mnew = fmaxf(m_r[r], mx);
      float alpha = __expf(m_r[r] - mnew);
      float e0 = __expf(p0[r] - mnew);
      float e1 = __expf(p1[r] - mnew);
      float rs = e0 + e1;
      #pragma unroll
      for (int off = 8; off >= 1; off >>= 1) rs += __shfl_xor(rs, off, 32);
      l_r[r] = l_r[r] * alpha + rs;
      m_r[r] = mnew;
      o0[r] *= alpha; o1[r] *= alpha; o2[r] *= alpha; o3[r] *= alpha;
      p0[r] = e0; p1[r] = e1;
    }
    // ---- stage P (C-layout) into LDS, read back as A-fragment ----
    #pragma unroll
    for (int r = 0; r < 8; ++r) {
      int rowl = r + rbase;
      pbuf[rowl * 32 + laneN]      = (_Float16)p0[r];
      pbuf[rowl * 32 + 16 + laneN] = (_Float16)p1[r];
    }
    __builtin_amdgcn_wave_barrier();              // same-wave LDS is in-order
    v16h pa = load_frag(pbuf + laneN * 32 + kb);
    // ---- O += P @ V ----
    o0 = wmma_f16(pa, vb0, o0);
    o1 = wmma_f16(pa, vb1, o1);
    o2 = wmma_f16(pa, vb2, o2);
    o3 = wmma_f16(pa, vb3, o3);
    __builtin_amdgcn_wave_barrier();
  }
  // ---- finalize ----
  v8f os[4] = {o0, o1, o2, o3};
  #pragma unroll
  for (int j = 0; j < 4; ++j) {
    #pragma unroll
    for (int r = 0; r < 8; ++r) {
      int t = t0 + r + rbase;
      float val = os[j][r] / l_r[r];
      Yh[((size_t)b * T_ + t) * HD_ + h * D_ + j * 16 + laneN] = (_Float16)val;
    }
  }
}

// ---------------- projection GEMM: Yh[4096,768] @ WprojT + b -> y (f32); 32x64 tile, ping-pong ----------------
__global__ __launch_bounds__(256) void proj_gemm_kernel(const _Float16* __restrict__ A,
                                                        const _Float16* __restrict__ WT,
                                                        const float* __restrict__ bias,
                                                        float* __restrict__ Y) {
  const int NT = C_ / 64;                         // 12
  int gwave = (blockIdx.x * 256 + threadIdx.x) >> 5;
  int lane  = threadIdx.x & 31;
  int mt = gwave / NT, nt = gwave % NT;
  if (mt >= (B_ * T_) / 32) return;
  int m0 = mt * 32, n0 = nt * 64;
  int laneN = lane & 15;
  int kb    = (lane >> 4) << 3;
  const _Float16* pa0 = A + (size_t)(m0 + laneN) * HD_ + kb;
  const _Float16* pa1 = A + (size_t)(m0 + 16 + laneN) * HD_ + kb;
  const _Float16* pb0 = WT + (size_t)(n0 + 0  + laneN) * HD_ + kb;
  const _Float16* pb1 = WT + (size_t)(n0 + 16 + laneN) * HD_ + kb;
  const _Float16* pb2 = WT + (size_t)(n0 + 32 + laneN) * HD_ + kb;
  const _Float16* pb3 = WT + (size_t)(n0 + 48 + laneN) * HD_ + kb;
  v8f acc[8];
  #pragma unroll
  for (int i = 0; i < 8; ++i) acc[i] = zero8();
  v16h xa0 = load_frag(pa0), xa1 = load_frag(pa1);
  v16h xb0 = load_frag(pb0), xb1 = load_frag(pb1);
  v16h xb2 = load_frag(pb2), xb3 = load_frag(pb3);
  #pragma unroll 1
  for (int k = 0; k < HD_; k += 64) {
    int k1 = k + 32;
    __builtin_prefetch(pa0 + k + 128, 0, 3);
    __builtin_prefetch(pa1 + k + 128, 0, 3);
    __builtin_prefetch(pb0 + k + 128, 0, 3);
    __builtin_prefetch(pb1 + k + 128, 0, 3);
    __builtin_prefetch(pb2 + k + 128, 0, 3);
    __builtin_prefetch(pb3 + k + 128, 0, 3);
    v16h ya0 = load_frag(pa0 + k1), ya1 = load_frag(pa1 + k1);
    v16h yb0 = load_frag(pb0 + k1), yb1 = load_frag(pb1 + k1);
    v16h yb2 = load_frag(pb2 + k1), yb3 = load_frag(pb3 + k1);
    GEMM8(xa0, xa1, xb0, xb1, xb2, xb3);
    if (k + 64 < HD_) {
      int k2 = k + 64;
      xa0 = load_frag(pa0 + k2); xa1 = load_frag(pa1 + k2);
      xb0 = load_frag(pb0 + k2); xb1 = load_frag(pb1 + k2);
      xb2 = load_frag(pb2 + k2); xb3 = load_frag(pb3 + k2);
    }
    GEMM8(ya0, ya1, yb0, yb1, yb2, yb3);
  }
  int rbase = kb;
  #pragma unroll
  for (int half = 0; half < 2; ++half) {
    #pragma unroll
    for (int j = 0; j < 4; ++j) {
      int n = n0 + j * 16 + laneN;
      float bn = bias[n];
      #pragma unroll
      for (int r = 0; r < 8; ++r) {
        int m = m0 + half * 16 + r + rbase;
        Y[(size_t)m * C_ + n] = acc[half * 4 + j][r] + bn;
      }
    }
  }
}

// ---------------- host launch ----------------
extern "C" void kernel_launch(void* const* d_in, const int* in_sizes, int n_in,
                              void* d_out, int out_size, void* d_ws, size_t ws_size,
                              hipStream_t stream) {
  const float* x      = (const float*)d_in[0];   // [B,T,C]
  const float* W_attn = (const float*)d_in[1];   // [C, 3*HD]
  const float* b_attn = (const float*)d_in[2];   // [3*HD]
  const float* W_proj = (const float*)d_in[3];   // [HD, C]
  const float* b_proj = (const float*)d_in[4];   // [C]
  float* y_out = (float*)d_out;                               // [B,T,C]
  float* m_out = (float*)d_out + (size_t)B_ * T_ * C_;        // [1,B,T,T]

  char* ws = (char*)d_ws;
  size_t off = 0;
  auto alloc = [&](size_t bytes) { char* p = ws + off; off += (bytes + 255) & ~(size_t)255; return p; };
  _Float16* x16  = (_Float16*)alloc((size_t)B_ * T_ * C_ * 2);
  _Float16* waT  = (_Float16*)alloc((size_t)N3_ * C_ * 2);
  _Float16* wpT  = (_Float16*)alloc((size_t)C_ * HD_ * 2);
  _Float16* q16  = (_Float16*)alloc((size_t)B_ * H_ * T_ * D_ * 2);
  _Float16* k16  = (_Float16*)alloc((size_t)B_ * H_ * T_ * D_ * 2);
  _Float16* vT16 = (_Float16*)alloc((size_t)B_ * H_ * T_ * D_ * 2);
  float*    Sff  = (float*)   alloc((size_t)B_ * T_ * T_ * 4);
  float*    part = (float*)   alloc((size_t)B_ * CH_ * T_ * 4);
  float*    fsum = (float*)   alloc((size_t)B_ * T_ * 4);
  _Float16* yh16 = (_Float16*)alloc((size_t)B_ * T_ * HD_ * 2);

  // prep conversions
  {
    int n = B_ * T_ * C_;
    cvt_f16_kernel<<<(n + 255) / 256, 256, 0, stream>>>(x, x16, n);
  }
  {
    int n = C_ * N3_;
    cvt_t_kernel<<<(n + 255) / 256, 256, 0, stream>>>(W_attn, waT, C_, N3_);
  }
  {
    int n = HD_ * C_;
    cvt_t_kernel<<<(n + 255) / 256, 256, 0, stream>>>(W_proj, wpT, HD_, C_);
  }
  // qkv GEMM: 128 m-tiles * 36 n-tiles = 4608 waves -> 576 blocks
  qkv_gemm_kernel<<<((B_ * T_ / 32) * (N3_ / 64) + 7) / 8, 256, 0, stream>>>(
      x16, waT, b_attn, q16, k16, vT16);
  // head-0 scores: B*128*32 wave-groups = 8192 waves -> 1024 blocks
  s0_kernel<<<(B_ * (T_ / 16) * (T_ / 64) + 7) / 8, 256, 0, stream>>>(q16, k16, Sff);
  // 3-phase column scan (in-place -> FF)
  scan_phase1_kernel<<<(B_ * CH_ * T_ + 255) / 256, 256, 0, stream>>>(Sff, part);
  scan_phase2_kernel<<<(B_ * T_ + 255) / 256, 256, 0, stream>>>(part);
  scan_phase3_kernel<<<(B_ * CH_ * T_ + 255) / 256, 256, 0, stream>>>(Sff, part);
  // clipped row sums
  ffsum_kernel<<<B_ * T_, 256, 0, stream>>>(Sff, fsum);
  // M output
  {
    size_t n = (size_t)B_ * T_ * T_;
    m_kernel<<<(unsigned)((n + 255) / 256), 256, 0, stream>>>(fsum, m_out);
  }
  // flash attention: B*H*128 = 3072 waves -> 384 blocks
  attn_kernel<<<(B_ * H_ * (T_ / 16) + 7) / 8, 256, 0, stream>>>(q16, k16, vT16, Sff, yh16);
  // output projection: 128*12 = 1536 waves -> 192 blocks
  proj_gemm_kernel<<<((B_ * T_ / 32) * (C_ / 64) + 7) / 8, 256, 0, stream>>>(
      yh16, wpT, b_proj, y_out);
}
